// SharedTransformerBlock_37409165148453
// MI455X (gfx1250) — matrix operand order repaired and checked by reading, not compiled, not run on hardware
//
#include <hip/hip_runtime.h>
#include <hip/hip_bf16.h>
#include <math.h>

// ---------------------------------------------------------------------------
// SharedTransformerBlock forward for MI455X (gfx1250, wave32, WMMA).
// GEMMs + attention contractions on v_wmma_f32_16x16x32_bf16; tile fills via
// global_load_async_to_lds_b128 (ASYNCcnt) where layout permits.
// ---------------------------------------------------------------------------

#define NHEADS 12
#define HDIM 64
#define DMODEL 768

typedef __bf16 bf16;
typedef __attribute__((ext_vector_type(16))) __bf16 v16bf;
typedef __attribute__((ext_vector_type(2)))  __bf16 v2bf;
typedef __attribute__((ext_vector_type(8)))  float  v8f;

__device__ __forceinline__ v8f wmma_bf16(v16bf a, v16bf b, v8f c) {
  return __builtin_amdgcn_wmma_f32_16x16x32_bf16(false, a, false, b, (short)0, c,
                                                 false, false);
}

// CDNA5 async global->LDS copy (16B per lane), tracked by ASYNCcnt.
__device__ __forceinline__ void async_ld_b128(unsigned lds_byte_off, const void* g) {
  asm volatile("global_load_async_to_lds_b128 %0, %1, off"
               :: "v"(lds_byte_off), "v"(g) : "memory");
}
__device__ __forceinline__ void wait_async0() {
  asm volatile("s_wait_asynccnt 0x0" ::: "memory");
}
__device__ __forceinline__ unsigned lds_off(const void* p) {
  return (unsigned)(size_t)p;  // flat shared addr: low 32 bits = LDS offset
}

// A fragment (16x32, MxK) from row-major bf16; p -> (row0,k0); ld elems.
// Lane m=lane&15, half=lane>>4; VGPR j holds K = 2(j&3)+8*half+16(j>>2), +1.
__device__ __forceinline__ v16bf frag_a(const bf16* p, int ld, int lane) {
  int m = lane & 15, hf = lane >> 4;
  const bf16* row = p + m * ld;
  v16bf f;
#pragma unroll
  for (int j = 0; j < 8; ++j) {
    int k = 2 * (j & 3) + 8 * hf + 16 * (j >> 2);
    v2bf pr = *(const v2bf*)(row + k);
    f[2 * j] = pr[0];
    f[2 * j + 1] = pr[1];
  }
  return f;
}

// B fragment (32x16) as M^T of row-major M (16 n-rows x 32 k-cols); pairs
// contiguous. Lane n=lane&15, half=lane>>4; VGPR j holds K = 2(j&3)+8*half+16(j>>2), +1.
__device__ __forceinline__ v16bf frag_bt(const bf16* p, int ld, int lane) {
  int n = lane & 15, hf = lane >> 4;
  const bf16* row = p + n * ld;
  v16bf f;
#pragma unroll
  for (int j = 0; j < 8; ++j) {
    int k = 2 * (j & 3) + 8 * hf + 16 * (j >> 2);
    v2bf pr = *(const v2bf*)(row + k);
    f[2 * j] = pr[0];
    f[2 * j + 1] = pr[1];
  }
  return f;
}

// ---------------------------------------------------------------------------
// GEMM: out[MxN] = epilogue(A[MxK]bf16 @ B[KxN]bf16)
// EPI: 0 none, 1 +bias, 2 +bias+GELU(exact), 3 +bias+residual
// ---------------------------------------------------------------------------
template <int EPI, bool GUARD, typename OutT>
__device__ __forceinline__ void epi_store(v8f (&acc)[2][2],
                                          const float* __restrict__ bias,
                                          const float* __restrict__ res,
                                          OutT* __restrict__ out, int M, int N,
                                          int row_base, int col_base, int lane) {
  int hf = lane >> 4, nn = lane & 15;
#pragma unroll
  for (int mt = 0; mt < 2; ++mt)
#pragma unroll
    for (int nt = 0; nt < 2; ++nt)
#pragma unroll
      for (int i = 0; i < 8; ++i) {
        int r = row_base + mt * 16 + i + 8 * hf;
        int c = col_base + nt * 16 + nn;
        if (!GUARD || r < M) {
          float v = acc[mt][nt][i];
          if (EPI >= 1) v += bias[c];
          if (EPI == 2) v = 0.5f * v * (1.0f + erff(v * 0.70710678118654752f));
          if (EPI == 3) v += res[(size_t)r * N + c];
          out[(size_t)r * N + c] = (OutT)v;
        }
      }
}

template <int EPI, typename OutT>
__global__ __launch_bounds__(256) void gemm_kernel(
    const bf16* __restrict__ A, const bf16* __restrict__ B,
    const float* __restrict__ bias, const float* __restrict__ res,
    OutT* __restrict__ out, int M, int N, int K) {
  __shared__ bf16 As[128 * 40];  // 128 x 32, 80B rows (16B-aligned, 20*m banks)
  __shared__ bf16 BsT[64 * 34];  // transposed tile: 64 n-rows x 32 k, 68B rows

  const int tid  = threadIdx.x;
  const int lane = tid & 31;
  const int wv   = tid >> 5;
  const int wm   = wv & 3;
  const int wn   = wv >> 2;
  const int row0 = blockIdx.y * 128;
  const int col0 = blockIdx.x * 64;
  const unsigned as_base = lds_off(As);

  v8f acc[2][2];
#pragma unroll
  for (int a = 0; a < 2; ++a)
#pragma unroll
    for (int b = 0; b < 2; ++b) acc[a][b] = {};

  const int ar = tid >> 1, ac = (tid & 1) * 16;   // A tile: 2 thr/row, 16 elems
  int agr = row0 + ar;
  if (agr >= M) agr = M - 1;  // clamp: dead rows never stored
  const int br = tid >> 3, bc = (tid & 7) * 8;    // B tile: 8 thr/row, 8 elems

  for (int k0 = 0; k0 < K; k0 += 32) {
    __syncthreads();
    {  // A tile: async global->LDS, two b128 per thread
      const bf16* g = A + (size_t)agr * K + k0 + ac;
      unsigned l = as_base + (unsigned)((ar * 40 + ac) * 2);
      async_ld_b128(l, g);
      async_ld_b128(l + 16, g + 8);
    }
    {  // B tile: load 8 n-consecutive, scatter-store transposed
      uint4 x = *(const uint4*)(B + (size_t)(k0 + br) * N + col0 + bc);
      bf16 tmp[8];
      *(uint4*)tmp = x;
#pragma unroll
      for (int j = 0; j < 8; ++j) BsT[(bc + j) * 34 + br] = tmp[j];
    }
    if (k0 + 32 < K) {  // warm L2 for next tiles (global_prefetch_b8)
      __builtin_prefetch(A + (size_t)agr * K + k0 + 32 + ac, 0, 1);
      __builtin_prefetch(B + (size_t)(k0 + 32 + br) * N + col0 + bc, 0, 1);
    }
    wait_async0();
    __syncthreads();

#pragma unroll
    for (int mt = 0; mt < 2; ++mt) {
      v16bf af = frag_a(As + (wm * 32 + mt * 16) * 40, 40, lane);
#pragma unroll
      for (int nt = 0; nt < 2; ++nt) {
        v16bf bf = frag_bt(BsT + (wn * 32 + nt * 16) * 34, 34, lane);
        acc[mt][nt] = wmma_bf16(af, bf, acc[mt][nt]);
      }
    }
  }

  if (row0 + 128 <= M)
    epi_store<EPI, false>(acc, bias, res, out, M, N, row0 + wm * 32,
                          col0 + wn * 32, lane);
  else
    epi_store<EPI, true>(acc, bias, res, out, M, N, row0 + wm * 32,
                         col0 + wn * 32, lane);
}

// ---------------------------------------------------------------------------
// Fused flash attention. Q/K/V: [(seq*NH+h)*S + tok]*64 + d (bf16, RoPE'd)
// O: [(seq*S+tok)*768 + h*64 + d] (bf16, token-major)
// ---------------------------------------------------------------------------
template <int S>
__global__ __launch_bounds__(32 * (S / 32)) void attn_kernel(
    const bf16* __restrict__ Q, const bf16* __restrict__ K,
    const bf16* __restrict__ V, bf16* __restrict__ O) {
  constexpr int NW    = S / 32;
  constexpr int CHUNK = (S >= 128) ? 128 : S;
  constexpr int LDK   = 72;         // 144B rows: 16B-aligned, conflict-free
  constexpr int LDV   = CHUNK + 2;  // transposed V rows, 4B-aligned
  __shared__ bf16 Ks[CHUNK * LDK];
  __shared__ bf16 VsT[HDIM * LDV];
  __shared__ bf16 Ps[NW * 2 * 16 * 34];

  const int tid  = threadIdx.x;
  const int lane = tid & 31;
  const int wv   = tid >> 5;
  const int seqh = blockIdx.x;
  const int seq = seqh / NHEADS, h = seqh % NHEADS;
  const bf16* Qb = Q + (size_t)seqh * S * HDIM;
  const bf16* Kb = K + (size_t)seqh * S * HDIM;
  const bf16* Vb = V + (size_t)seqh * S * HDIM;
  const unsigned ks_base = lds_off(Ks);

  v16bf qf[2][2];
#pragma unroll
  for (int mt = 0; mt < 2; ++mt)
#pragma unroll
    for (int kf = 0; kf < 2; ++kf)
      qf[mt][kf] = frag_a(Qb + (wv * 32 + mt * 16) * HDIM + kf * 32, HDIM, lane);

  v8f oacc[2][4];
  float mrun[2][8], lrun[2][8];
#pragma unroll
  for (int mt = 0; mt < 2; ++mt) {
#pragma unroll
    for (int nt = 0; nt < 4; ++nt) oacc[mt][nt] = {};
#pragma unroll
    for (int i = 0; i < 8; ++i) { mrun[mt][i] = -3.0e38f; lrun[mt][i] = 0.f; }
  }

  bf16* Pw = Ps + wv * (2 * 16 * 34);

  for (int c0 = 0; c0 < S; c0 += CHUNK) {
    __syncthreads();
    // K chunk: async global->LDS (row-major, padded rows)
    for (int u = tid; u < CHUNK * 8; u += 32 * NW) {
      int tok = u >> 3, kc = (u & 7) * 8;
      async_ld_b128(ks_base + (unsigned)((tok * LDK + kc) * 2),
                    Kb + (size_t)(c0 + tok) * HDIM + kc);
    }
    // V chunk: transposed store (fragment-friendly: contiguous k-pairs)
    for (int u = tid; u < CHUNK * 8; u += 32 * NW) {
      int tok = u >> 3, kc = (u & 7) * 8;
      uint4 x = *(const uint4*)(Vb + (size_t)(c0 + tok) * HDIM + kc);
      bf16 tmp[8];
      *(uint4*)tmp = x;
#pragma unroll
      for (int j = 0; j < 8; ++j) VsT[(kc + j) * LDV + tok] = tmp[j];
    }
    wait_async0();
    __syncthreads();

    for (int t0 = 0; t0 < CHUNK; t0 += 32) {
      v16bf kfr[2][2];  // K^T fragments [khalf][ntile]
#pragma unroll
      for (int nt = 0; nt < 2; ++nt)
#pragma unroll
        for (int kf = 0; kf < 2; ++kf)
          kfr[kf][nt] = frag_bt(Ks + (t0 + nt * 16) * LDK + kf * 32, LDK, lane);

#pragma unroll
      for (int mt = 0; mt < 2; ++mt) {
        v8f c0v = {}, c1v = {};
        c0v = wmma_bf16(qf[mt][0], kfr[0][0], c0v);
        c0v = wmma_bf16(qf[mt][1], kfr[1][0], c0v);
        c1v = wmma_bf16(qf[mt][0], kfr[0][1], c1v);
        c1v = wmma_bf16(qf[mt][1], kfr[1][1], c1v);

        float p0[8], p1[8], sc[8];
#pragma unroll
        for (int i = 0; i < 8; ++i) {
          float a = c0v[i] * 0.125f, b = c1v[i] * 0.125f;  // HD^-0.5
          float t = fmaxf(a, b);
#pragma unroll
          for (int msk = 1; msk < 16; msk <<= 1)
            t = fmaxf(t, __shfl_xor(t, msk, 32));  // within 16-lane half
          float mn = fmaxf(mrun[mt][i], t);
          sc[i] = __expf(mrun[mt][i] - mn);
          mrun[mt][i] = mn;
          p0[i] = __expf(a - mn);
          p1[i] = __expf(b - mn);
          float rs = p0[i] + p1[i];
#pragma unroll
          for (int msk = 1; msk < 16; msk <<= 1) rs += __shfl_xor(rs, msk, 32);
          lrun[mt][i] = lrun[mt][i] * sc[i] + rs;
        }
#pragma unroll
        for (int nt = 0; nt < 4; ++nt)
#pragma unroll
          for (int i = 0; i < 8; ++i) oacc[mt][nt][i] *= sc[i];

        {  // P tile -> wave-private LDS (WMMA A layout round-trip)
          int hf = lane >> 4, nn = lane & 15;
          bf16* pw = Pw + mt * 16 * 34;
#pragma unroll
          for (int i = 0; i < 8; ++i) {
            pw[(i + 8 * hf) * 34 + nn]      = (bf16)p0[i];
            pw[(i + 8 * hf) * 34 + 16 + nn] = (bf16)p1[i];
          }
        }
      }
      v16bf pa[2];
#pragma unroll
      for (int mt = 0; mt < 2; ++mt) pa[mt] = frag_a(Pw + mt * 16 * 34, 34, lane);
#pragma unroll
      for (int nt = 0; nt < 4; ++nt) {
        v16bf vf = frag_bt(VsT + (nt * 16) * LDV + t0, LDV, lane);
        oacc[0][nt] = wmma_bf16(pa[0], vf, oacc[0][nt]);
        oacc[1][nt] = wmma_bf16(pa[1], vf, oacc[1][nt]);
      }
    }
  }

  int hf = lane >> 4, nn = lane & 15;
#pragma unroll
  for (int mt = 0; mt < 2; ++mt)
#pragma unroll
    for (int nt = 0; nt < 4; ++nt)
#pragma unroll
      for (int i = 0; i < 8; ++i) {
        int row = wv * 32 + mt * 16 + i + 8 * hf;
        int col = nt * 16 + nn;
        float val = oacc[mt][nt][i] / lrun[mt][i];
        O[((size_t)(seq * S + row)) * DMODEL + h * HDIM + col] = (bf16)val;
      }
}

// ---------------------------------------------------------------------------
// LayerNorm (row of 768) -> bf16
// ---------------------------------------------------------------------------
__device__ __forceinline__ void ln_row_dev(const float* __restrict__ x,
                                           bf16* __restrict__ y,
                                           const float* __restrict__ w,
                                           const float* __restrict__ b) {
  int tid = threadIdx.x;
  float v0 = x[tid], v1 = x[tid + 256], v2 = x[tid + 512];
  float s = v0 + v1 + v2;
  float q = v0 * v0 + v1 * v1 + v2 * v2;
#pragma unroll
  for (int m = 1; m < 32; m <<= 1) {
    s += __shfl_xor(s, m, 32);
    q += __shfl_xor(q, m, 32);
  }
  __shared__ float ss[8], qq[8];
  if ((tid & 31) == 0) { ss[tid >> 5] = s; qq[tid >> 5] = q; }
  __syncthreads();
  s = 0.f; q = 0.f;
#pragma unroll
  for (int j = 0; j < 8; ++j) { s += ss[j]; q += qq[j]; }
  float mean = s * (1.f / 768.f);
  float var  = q * (1.f / 768.f) - mean * mean;
  float inv  = rsqrtf(var + 1e-5f);
  y[tid]       = (bf16)((v0 - mean) * inv * w[tid] + b[tid]);
  y[tid + 256] = (bf16)((v1 - mean) * inv * w[tid + 256] + b[tid + 256]);
  y[tid + 512] = (bf16)((v2 - mean) * inv * w[tid + 512] + b[tid + 512]);
}

__global__ __launch_bounds__(256) void ln_kernel(const float* __restrict__ X,
                                                 bf16* __restrict__ Y,
                                                 const float* __restrict__ w,
                                                 const float* __restrict__ b) {
  ln_row_dev(X + (size_t)blockIdx.x * DMODEL, Y + (size_t)blockIdx.x * DMODEL, w, b);
}

// Gather temporal token stream [vp | vf | ap | af] per (b,s) and apply LN3.
__global__ __launch_bounds__(256) void gather_ln3_kernel(
    const float* __restrict__ xvp, const float* __restrict__ xvf,
    const float* __restrict__ xap, const float* __restrict__ xaf,
    bf16* __restrict__ Y, const float* __restrict__ w,
    const float* __restrict__ b) {
  int r  = blockIdx.x;  // bs*32 + tt
  int bs = r >> 5, tt = r & 31;
  int bb = bs >> 8, sp = bs & 255;
  const float* x;
  if (tt < 8)       x = xvp + ((size_t)((bb * 8 + tt) * 256 + sp)) * DMODEL;
  else if (tt < 16) x = xvf + ((size_t)((bb * 8 + (tt - 8)) * 256 + sp)) * DMODEL;
  else if (tt < 24) x = xap + (size_t)(bb * 8 + (tt - 16)) * DMODEL;
  else              x = xaf + (size_t)(bb * 8 + (tt - 24)) * DMODEL;
  ln_row_dev(x, Y + (size_t)r * DMODEL, w, b);
}

// ---------------------------------------------------------------------------
// RoPE + relayout to (seq,head,tok,hd) bf16. One thread per (token, head).
// ---------------------------------------------------------------------------
__global__ void rope2d_kernel(const float* __restrict__ qkv,
                              bf16* __restrict__ Qo, bf16* __restrict__ Ko,
                              bf16* __restrict__ Vo) {
  int idx = blockIdx.x * blockDim.x + threadIdx.x;
  if (idx >= 16384 * NHEADS) return;
  int h = idx % NHEADS, tok = idx / NHEADS;
  int bt = tok >> 8, s = tok & 255;
  int hp = s >> 4, wp = s & 15;
  const float* q = qkv + (size_t)tok * 2304 + h * HDIM;
  const float* k = q + DMODEL;
  const float* v = q + 2 * DMODEL;
  size_t dst = ((size_t)(bt * NHEADS + h) * 256 + s) * HDIM;
#pragma unroll
  for (int half = 0; half < 2; ++half) {
    float pos = half ? (float)wp : (float)hp;
    int base = half * 32;
    for (int i = 0; i < 16; ++i) {
      float f = __powf(10000.f, -(float)(2 * i) / 32.f);
      float c, sn;
      __sincosf(pos * f, &sn, &c);
      float q0 = q[base + i], q1 = q[base + i + 16];
      float k0 = k[base + i], k1 = k[base + i + 16];
      Qo[dst + base + i]      = (bf16)(q0 * c - q1 * sn);
      Qo[dst + base + i + 16] = (bf16)(q1 * c + q0 * sn);
      Ko[dst + base + i]      = (bf16)(k0 * c - k1 * sn);
      Ko[dst + base + i + 16] = (bf16)(k1 * c + k0 * sn);
    }
  }
  for (int d = 0; d < HDIM; ++d) Vo[dst + d] = (bf16)v[d];
}

__global__ void rope1d_kernel(const float* __restrict__ qkv,
                              bf16* __restrict__ Qo, bf16* __restrict__ Ko,
                              bf16* __restrict__ Vo) {
  int idx = blockIdx.x * blockDim.x + threadIdx.x;
  if (idx >= 32768 * NHEADS) return;
  int h = idx % NHEADS, tok = idx / NHEADS;
  int bs = tok >> 5, tt = tok & 31;
  const float* q = qkv + (size_t)tok * 2304 + h * HDIM;
  const float* k = q + DMODEL;
  const float* v = q + 2 * DMODEL;
  size_t dst = ((size_t)(bs * NHEADS + h) * 32 + tt) * HDIM;
  float pos = (float)tt;
  for (int i = 0; i < 32; ++i) {
    float f = __powf(10000.f, -(float)(2 * i) / 64.f);
    float c, sn;
    __sincosf(pos * f, &sn, &c);
    float q0 = q[i], q1 = q[i + 32];
    float k0 = k[i], k1 = k[i + 32];
    Qo[dst + i]      = (bf16)(q0 * c - q1 * sn);
    Qo[dst + i + 32] = (bf16)(q1 * c + q0 * sn);
    Ko[dst + i]      = (bf16)(k0 * c - k1 * sn);
    Ko[dst + i + 32] = (bf16)(k1 * c + k0 * sn);
  }
  for (int d = 0; d < HDIM; ++d) Vo[dst + d] = (bf16)v[d];
}

// Scatter temporal-attn video deltas back into the residual streams.
__global__ void scatter_vid_kernel(const float* __restrict__ ot,
                                   float* __restrict__ xvp,
                                   float* __restrict__ xvf) {
  size_t idx = (size_t)blockIdx.x * blockDim.x + threadIdx.x;
  if (idx >= (size_t)2 * 4 * 8 * 256 * DMODEL) return;
  int c = (int)(idx % DMODEL);
  size_t r = idx / DMODEL;
  int half = (int)(r / 8192);
  int r2 = (int)(r % 8192);
  int bb = r2 / 2048;
  int t  = (r2 / 256) % 8;
  int sp = r2 % 256;
  int tt = half * 8 + t;
  int bs = bb * 256 + sp;
  float d = ot[((size_t)bs * 32 + tt) * DMODEL + c];
  float* x = half ? xvf : xvp;
  x[((size_t)((bb * 8 + t) * 256 + sp)) * DMODEL + c] += d;
}

// Audio delta = mean over spatial positions of its temporal-attn tokens.
__global__ void audio_mean_kernel(const float* __restrict__ ot,
                                  float* __restrict__ xa, int ttoff) {
  int idx = blockIdx.x * blockDim.x + threadIdx.x;
  if (idx >= 4 * 8 * DMODEL) return;
  int c = idx % DMODEL;
  int j = (idx / DMODEL) % 8;
  int bb = idx / (DMODEL * 8);
  float s = 0.f;
  for (int sp = 0; sp < 256; ++sp)
    s += ot[((size_t)(bb * 256 + sp) * 32 + ttoff + j) * DMODEL + c];
  xa[(size_t)(bb * 8 + j) * DMODEL + c] += s * (1.f / 256.f);
}

__global__ void cvt_kernel(const float* __restrict__ s, bf16* __restrict__ d, int n) {
  int i = blockIdx.x * blockDim.x + threadIdx.x;
  if (i < n) d[i] = (bf16)s[i];
}

// ---------------------------------------------------------------------------
// Host orchestration
// ---------------------------------------------------------------------------
extern "C" void kernel_launch(void* const* d_in, const int* in_sizes, int n_in,
                              void* d_out, int out_size, void* d_ws,
                              size_t ws_size, hipStream_t stream) {
  (void)in_sizes; (void)n_in; (void)out_size; (void)ws_size;
  const int VTOK = 16384;  // video tokens (vp + vf)
  const int TTOK = 32768;  // temporal tokens (1024 seqs * 32)

  float* x_vp = (float*)d_out;  // (4,8,16,16,768)
  float* x_vf = x_vp + 6291456;
  float* x_ap = x_vf + 6291456;  // (4,8,768)
  float* x_af = x_ap + 24576;

  hipMemcpyAsync(x_vp, d_in[0], 6291456 * 4, hipMemcpyDeviceToDevice, stream);
  hipMemcpyAsync(x_vf, d_in[1], 6291456 * 4, hipMemcpyDeviceToDevice, stream);
  hipMemcpyAsync(x_ap, d_in[2], 24576 * 4, hipMemcpyDeviceToDevice, stream);
  hipMemcpyAsync(x_af, d_in[3], 24576 * 4, hipMemcpyDeviceToDevice, stream);

  char* ws = (char*)d_ws;
  size_t off = 0;
  auto alloc = [&](size_t bytes) -> void* {
    void* p = ws + off;
    off = (off + bytes + 255) & ~(size_t)255;
    return p;
  };

  bf16* w_saqkv  = (bf16*)alloc((size_t)768 * 2304 * 2);
  bf16* w_saproj = (bf16*)alloc((size_t)768 * 768 * 2);
  bf16* w_smlp1  = (bf16*)alloc((size_t)768 * 3072 * 2);
  bf16* w_smlp2  = (bf16*)alloc((size_t)3072 * 768 * 2);
  bf16* w_taqkv  = (bf16*)alloc((size_t)768 * 2304 * 2);
  bf16* w_taproj = (bf16*)alloc((size_t)768 * 768 * 2);
  bf16* w_f1[4];
  bf16* w_f2[4];
  for (int i = 0; i < 4; ++i) {
    w_f1[i] = (bf16*)alloc((size_t)768 * 3072 * 2);
    w_f2[i] = (bf16*)alloc((size_t)3072 * 768 * 2);
  }

  bf16*  xln  = (bf16*)alloc((size_t)TTOK * DMODEL * 2);
  float* qkv  = (float*)alloc((size_t)TTOK * 2304 * 4);
  bf16*  Qb   = (bf16*)alloc((size_t)TTOK * DMODEL * 2);
  bf16*  Kb   = (bf16*)alloc((size_t)TTOK * DMODEL * 2);
  bf16*  Vb   = (bf16*)alloc((size_t)TTOK * DMODEL * 2);
  bf16*  attn = (bf16*)alloc((size_t)TTOK * DMODEL * 2);
  bf16*  hid  = (bf16*)alloc((size_t)VTOK * 3072 * 2);
  float* ot   = (float*)alloc((size_t)TTOK * DMODEL * 4);

  auto conv = [&](int idx, bf16* dst, int n) {
    cvt_kernel<<<(n + 255) / 256, 256, 0, stream>>>((const float*)d_in[idx], dst, n);
  };
  conv(4,  w_saqkv,  768 * 2304);
  conv(5,  w_saproj, 768 * 768);
  conv(7,  w_smlp1,  768 * 3072);
  conv(9,  w_smlp2,  3072 * 768);
  conv(11, w_taqkv,  768 * 2304);
  conv(12, w_taproj, 768 * 768);
  const int f_w1[4] = {22, 26, 30, 34};
  const int f_w2[4] = {24, 28, 32, 36};
  for (int i = 0; i < 4; ++i) {
    conv(f_w1[i], w_f1[i], 768 * 3072);
    conv(f_w2[i], w_f2[i], 3072 * 768);
  }

  const float* ln1w = (const float*)d_in[14]; const float* ln1b = (const float*)d_in[15];
  const float* ln2w = (const float*)d_in[16]; const float* ln2b = (const float*)d_in[17];
  const float* ln3w = (const float*)d_in[18]; const float* ln3b = (const float*)d_in[19];
  const float* ln4w = (const float*)d_in[20]; const float* ln4b = (const float*)d_in[21];
  const float* sa_proj_b = (const float*)d_in[6];
  const float* smlp_b1   = (const float*)d_in[8];
  const float* smlp_b2   = (const float*)d_in[10];
  const float* ta_proj_b = (const float*)d_in[13];

  // ---- video stage (vp and vf batched: contiguous 16384 rows) ----
  ln_kernel<<<VTOK, 256, 0, stream>>>(x_vp, xln, ln1w, ln1b);
  gemm_kernel<0, float><<<dim3(2304 / 64, VTOK / 128), 256, 0, stream>>>(
      xln, w_saqkv, nullptr, nullptr, qkv, VTOK, 2304, 768);
  rope2d_kernel<<<(VTOK * NHEADS + 255) / 256, 256, 0, stream>>>(qkv, Qb, Kb, Vb);
  attn_kernel<256><<<64 * NHEADS, 256, 0, stream>>>(Qb, Kb, Vb, attn);
  gemm_kernel<3, float><<<dim3(768 / 64, VTOK / 128), 256, 0, stream>>>(
      attn, w_saproj, sa_proj_b, x_vp, x_vp, VTOK, 768, 768);
  ln_kernel<<<VTOK, 256, 0, stream>>>(x_vp, xln, ln2w, ln2b);
  gemm_kernel<2, bf16><<<dim3(3072 / 64, VTOK / 128), 256, 0, stream>>>(
      xln, w_smlp1, smlp_b1, nullptr, hid, VTOK, 3072, 768);
  gemm_kernel<3, float><<<dim3(768 / 64, VTOK / 128), 256, 0, stream>>>(
      hid, w_smlp2, smlp_b2, x_vp, x_vp, VTOK, 768, 3072);

  // ---- temporal stage ----
  gather_ln3_kernel<<<TTOK, 256, 0, stream>>>(x_vp, x_vf, x_ap, x_af, xln, ln3w, ln3b);
  gemm_kernel<0, float><<<dim3(2304 / 64, TTOK / 128), 256, 0, stream>>>(
      xln, w_taqkv, nullptr, nullptr, qkv, TTOK, 2304, 768);
  rope1d_kernel<<<(TTOK * NHEADS + 255) / 256, 256, 0, stream>>>(qkv, Qb, Kb, Vb);
  attn_kernel<32><<<1024 * NHEADS, 32, 0, stream>>>(Qb, Kb, Vb, attn);
  gemm_kernel<1, float><<<dim3(768 / 64, TTOK / 128), 256, 0, stream>>>(
      attn, w_taproj, ta_proj_b, nullptr, ot, TTOK, 768, 768);
  scatter_vid_kernel<<<(int)(((size_t)2 * 4 * 8 * 256 * DMODEL + 255) / 256), 256, 0,
                       stream>>>(ot, x_vp, x_vf);
  audio_mean_kernel<<<(4 * 8 * DMODEL + 255) / 256, 256, 0, stream>>>(ot, x_ap, 16);
  audio_mean_kernel<<<(4 * 8 * DMODEL + 255) / 256, 256, 0, stream>>>(ot, x_af, 24);

  // ---- final per-stream MLPs ----
  auto final_mlp = [&](float* x, int rows, bf16* w1, int b1i, bf16* w2, int b2i) {
    const float* b1 = (const float*)d_in[b1i];
    const float* b2 = (const float*)d_in[b2i];
    ln_kernel<<<rows, 256, 0, stream>>>(x, xln, ln4w, ln4b);
    gemm_kernel<2, bf16><<<dim3(3072 / 64, (rows + 127) / 128), 256, 0, stream>>>(
        xln, w1, b1, nullptr, hid, rows, 3072, 768);
    gemm_kernel<3, float><<<dim3(768 / 64, (rows + 127) / 128), 256, 0, stream>>>(
        hid, w2, b2, x, x, rows, 768, 3072);
  };
  final_mlp(x_vp, 8192, w_f1[0], 23, w_f2[0], 25);
  final_mlp(x_vf, 8192, w_f1[1], 27, w_f2[1], 29);
  final_mlp(x_ap, 32,   w_f1[2], 31, w_f2[2], 33);
  final_mlp(x_af, 32,   w_f1[3], 35, w_f2[3], 37);
}
